// RegionalAttentionOp_87007447483110
// MI455X (gfx1250) — matrix operand order repaired and checked by reading, not compiled, not run on hardware
//
#include <hip/hip_runtime.h>

typedef __attribute__((ext_vector_type(16))) __bf16 v16bf;
typedef __attribute__((ext_vector_type(8)))  float  v8f;

#define HEADS   16
#define DHEAD   64
#define SEQ     7168
#define NKEYS   768          // 256 base + 2*256 regional
#define SCALE   0.125f       // 64^-0.5
#define NEGB    (-1e30f)

#define KS_STRIDE 72         // padded key row stride (bf16 elems)
#define VT_STRIDE 776        // padded Vt row stride (bf16 elems)
#define KS_BYTES  (NKEYS * KS_STRIDE * 2)   // 110592
#define VT_BYTES  (DHEAD * VT_STRIDE * 2)   //  99328
#define SMEM_BYTES (KS_BYTES + VT_BYTES)    // 209920 < 320KB WGP LDS

// workspace layout (bytes)
#define WS_KC 0                         // bf16 [16][768][64]
#define WS_VT (HEADS*NKEYS*DHEAD*2)     // bf16 [16][64][768]
#define WS_MB (2*HEADS*NKEYS*DHEAD*2)   // u32  [7168]

union V16u { v16bf v; uint4 u[2]; };

// ---------------------------------------------------------------------------
// Mask preprocessing: region_masks (2,121,32,56,1) -> per-query 3-bit group mask
// ---------------------------------------------------------------------------
__global__ void mask_kernel(const float* __restrict__ rm, unsigned* __restrict__ bits) {
    int s = blockIdx.x * blockDim.x + threadIdx.x;
    if (s >= SEQ) return;
    int t = s / 448, rem = s % 448, y = rem / 28, x = rem % 28;
    bool on[2];
#pragma unroll
    for (int r = 0; r < 2; ++r) {
        const float* M = rm + r * 121 * 32 * 56;
        float v;
        if (t == 0) {
            const float* F = M;  // frame 0
            v = 0.25f * (F[(2*y)*56 + 2*x]   + F[(2*y)*56 + 2*x + 1] +
                         F[(2*y+1)*56 + 2*x] + F[(2*y+1)*56 + 2*x + 1]);
        } else {
            const float* F3 = M + (8*t - 4) * 32 * 56;  // chunk mid frame 3
            const float* F4 = M + (8*t - 3) * 32 * 56;  // chunk mid frame 4
            float acc = 0.f;
#pragma unroll
            for (int dy = 0; dy < 2; ++dy)
#pragma unroll
                for (int dx = 0; dx < 2; ++dx) {
                    int hh = 2*y + dy, ww = 2*x + dx;
                    acc += 0.5f * (F3[hh*56 + ww] + F4[hh*56 + ww]);
                }
            v = 0.25f * acc;
        }
        on[r] = (v > 0.5f);
    }
    unsigned b = (unsigned)(!(on[0] || on[1])) | ((unsigned)on[0] << 1) | ((unsigned)on[1] << 2);
    bits[s] = b;
}

// ---------------------------------------------------------------------------
// Convert K/V (+regional) f32 -> bf16; K as [h][key][d], V transposed [h][d][key]
// ---------------------------------------------------------------------------
__global__ void convert_kernel(const float* __restrict__ k,  const float* __restrict__ v,
                               const float* __restrict__ rk, const float* __restrict__ rv,
                               __bf16* __restrict__ Kc, __bf16* __restrict__ Vt) {
    int i = blockIdx.x * blockDim.x + threadIdx.x;      // < 16*768*64
    int h   = i / (NKEYS * DHEAD);
    int rem = i % (NKEYS * DHEAD);
    int key = rem / DHEAD;
    int d   = rem % DHEAD;
    float kv, vv;
    if (key < 256) {
        size_t off = (size_t)key * 1024 + h * 64 + d;   // (P,B,H,D)
        kv = k[off]; vv = v[off];
    } else {
        size_t off = (size_t)(key - 256) * 1024 + h * 64 + d;  // (R*P,B,H,D) flat
        kv = rk[off]; vv = rv[off];
    }
    Kc[i] = (__bf16)kv;
    Vt[(size_t)h * DHEAD * NKEYS + (size_t)d * NKEYS + key] = (__bf16)vv;
}

// ---------------------------------------------------------------------------
// A-operand loader: 16x32 A tile, contiguous k order (lane m<16: cols {c..c+15},
// lane m+16: {c+16..c+31}) from a row-major padded buffer.
// ---------------------------------------------------------------------------
__device__ __forceinline__ v16bf load_a(const __bf16* base, int row0, int rstride,
                                        int col0, int lane) {
    int r  = row0 + (lane & 15);
    int cc = col0 + ((lane >> 4) << 4);
    const __bf16* p = base + r * rstride + cc;
    V16u t;
    t.u[0] = *(const uint4*)(p);
    t.u[1] = *(const uint4*)(p + 8);
    return t.v;
}

// Online-softmax step for one 32-key chunk: updates running (m, l), returns the
// packed P^T B-operand (32 keys x 16 queries) and the accumulator correction.
// Stats are per-query, held coherently in lane pair {q, q+16}.
__device__ __forceinline__ v16bf softmax_pb(const float* x0, const float* x1, float add,
                                            float& m, float& l, float& corr, int lane) {
    float tmax = x0[0] + add;
#pragma unroll
    for (int i = 1; i < 8; ++i) tmax = fmaxf(tmax, x0[i] + add);
#pragma unroll
    for (int i = 0; i < 8; ++i) tmax = fmaxf(tmax, x1[i] + add);
    tmax = fmaxf(tmax, __shfl_xor(tmax, 16, 32));        // share across lane pair
    float mnew = fmaxf(m, tmax);
    corr = __expf(m - mnew);
    m = mnew;

    float e0[8], e1[8], ssum = 0.f;
#pragma unroll
    for (int i = 0; i < 8; ++i) {
        e0[i] = __expf(x0[i] + add - mnew);
        e1[i] = __expf(x1[i] + add - mnew);
        ssum += e0[i] + e1[i];
    }
    l = l * corr + ssum;

    // lane q holds keys kb+0..7, kb+16..23; lane q+16 keys kb+8..15, kb+24..31: swap to make them contiguous.
    const bool lo = (lane < 16);
    v16bf pb;
#pragma unroll
    for (int i = 0; i < 8; ++i) {
        float send = lo ? e1[i] : e0[i];
        float recv = __shfl_xor(send, 16, 32);
        pb[i]     = (__bf16)(lo ? e0[i] : recv);
        pb[i + 8] = (__bf16)(lo ? recv  : e1[i]);
    }
    return pb;
}

// ---------------------------------------------------------------------------
// Attention: one head per blockIdx.y, 128 queries per block (8 waves x 16 q).
// ---------------------------------------------------------------------------
__global__ void __launch_bounds__(256, 1)
attn_kernel(const float* __restrict__ q, const __bf16* __restrict__ Kc,
            const __bf16* __restrict__ Vt, const unsigned* __restrict__ mbits,
            float* __restrict__ out) {
    extern __shared__ __align__(16) char smem[];
    __bf16* Ks  = (__bf16*)smem;                 // [768][72] padded
    __bf16* Vts = (__bf16*)(smem + KS_BYTES);    // [64][776] padded

    const int h   = blockIdx.y;
    const int tid = threadIdx.x;

    // Cooperative LDS fill (K and V^T for this head; ~200KB, reused by 8 waves)
    const __bf16* Kg = Kc + (size_t)h * NKEYS * DHEAD;
    const __bf16* Vg = Vt + (size_t)h * DHEAD * NKEYS;
    for (int idx = tid * 8; idx < NKEYS * DHEAD; idx += 256 * 8) {
        int key = idx >> 6, d = idx & 63;
        *(uint4*)&Ks[key * KS_STRIDE + d] = *(const uint4*)&Kg[idx];
    }
    for (int idx = tid * 8; idx < DHEAD * NKEYS; idx += 256 * 8) {
        int d = idx / NKEYS, kk = idx % NKEYS;
        *(uint4*)&Vts[d * VT_STRIDE + kk] = *(const uint4*)&Vg[idx];
    }
    __syncthreads();

    const int wave = tid >> 5;
    const int lane = tid & 31;
    const int qi   = lane & 15;
    const int hi   = lane >> 4;
    const int srow = blockIdx.x * 128 + wave * 16 + qi;

    // Q^T B-operands (32x16, d-chunks): lane q holds Q[q][c..c+15], lane q+16 [c+16..c+31]
    v16bf qb[2];
    {
        const float* qp = q + (size_t)srow * (HEADS * DHEAD) + h * DHEAD;
#pragma unroll
        for (int op = 0; op < 2; ++op) {
            int d0 = op * 32 + hi * 16;
            v16bf vv;
#pragma unroll
            for (int i = 0; i < 16; ++i) vv[i] = (__bf16)qp[d0 + i];
            qb[op] = vv;
        }
    }
    const unsigned mb = mbits[srow];

    v8f o_r[4] = {v8f{}, v8f{}, v8f{}, v8f{}};
    v8f o_b[4] = {v8f{}, v8f{}, v8f{}, v8f{}};
    float m_r = NEGB, l_r = 0.f;
    float m_b = NEGB, l_b = 0.f;

    for (int c = 0; c < 24; ++c) {                       // 32 keys per chunk
        const int kb = c * 32;
        // S^T = K * Q^T : two 16-key score tiles, reduce over d=64 (2 WMMAs each)
        v8f sa = {}, sb = {};
#pragma unroll
        for (int dc = 0; dc < 2; ++dc) {
            v16bf a0 = load_a(Ks, kb,      KS_STRIDE, dc * 32, lane);
            v16bf a1 = load_a(Ks, kb + 16, KS_STRIDE, dc * 32, lane);
            sa = __builtin_amdgcn_wmma_f32_16x16x32_bf16(false, a0, false, qb[dc],
                                                         (short)0, sa, false, false);
            sb = __builtin_amdgcn_wmma_f32_16x16x32_bf16(false, a1, false, qb[dc],
                                                         (short)0, sb, false, false);
        }
        float x0[8], x1[8];
#pragma unroll
        for (int i = 0; i < 8; ++i) { x0[i] = sa[i] * SCALE; x1[i] = sb[i] * SCALE; }

        // regional stream: per-query group bias (group = c/8; bit0=glob,1=rm0,2=rm1)
        const int g = c >> 3;
        const float add = ((mb >> g) & 1u) ? 0.f : NEGB;
        float corr_r, corr_b;
        v16bf pb_r = softmax_pb(x0, x1, add, m_r, l_r, corr_r, lane);
        v16bf pb_b = {};
        if (c < 8)                                        // base attention: first 256 keys
            pb_b = softmax_pb(x0, x1, 0.f, m_b, l_b, corr_b, lane);

        // O^T = V^T * P^T : share the 4 V^T A-tiles across both streams
#pragma unroll
        for (int t = 0; t < 4; ++t) {
            v16bf va = load_a(Vts, 16 * t, VT_STRIDE, kb, lane);
#pragma unroll
            for (int i = 0; i < 8; ++i) o_r[t][i] *= corr_r;
            o_r[t] = __builtin_amdgcn_wmma_f32_16x16x32_bf16(false, va, false, pb_r,
                                                             (short)0, o_r[t], false, false);
            if (c < 8) {
#pragma unroll
                for (int i = 0; i < 8; ++i) o_b[t][i] *= corr_b;
                o_b[t] = __builtin_amdgcn_wmma_f32_16x16x32_bf16(false, va, false, pb_b,
                                                                 (short)0, o_b[t], false, false);
            }
        }
    }

    // Combine lane-pair partial sums, rescale, blend 0.5/0.5, store.
    float lr = l_r + __shfl_xor(l_r, 16, 32);
    float lb = l_b + __shfl_xor(l_b, 16, 32);
    float ir = 0.5f / lr, ib = 0.5f / lb;
    float* op = out + (size_t)srow * (HEADS * DHEAD) + h * DHEAD + hi * 8;
#pragma unroll
    for (int t = 0; t < 4; ++t) {
        float4 a, b;
        a.x = o_r[t][0] * ir + o_b[t][0] * ib;
        a.y = o_r[t][1] * ir + o_b[t][1] * ib;
        a.z = o_r[t][2] * ir + o_b[t][2] * ib;
        a.w = o_r[t][3] * ir + o_b[t][3] * ib;
        b.x = o_r[t][4] * ir + o_b[t][4] * ib;
        b.y = o_r[t][5] * ir + o_b[t][5] * ib;
        b.z = o_r[t][6] * ir + o_b[t][6] * ib;
        b.w = o_r[t][7] * ir + o_b[t][7] * ib;
        *(float4*)(op + t * 16)     = a;
        *(float4*)(op + t * 16 + 4) = b;
    }
}

// ---------------------------------------------------------------------------
extern "C" void kernel_launch(void* const* d_in, const int* in_sizes, int n_in,
                              void* d_out, int out_size, void* d_ws, size_t ws_size,
                              hipStream_t stream) {
    const float* q  = (const float*)d_in[0];
    const float* k  = (const float*)d_in[1];
    const float* v  = (const float*)d_in[2];
    const float* rk = (const float*)d_in[3];
    const float* rv = (const float*)d_in[4];
    const float* rm = (const float*)d_in[5];

    __bf16*   Kc = (__bf16*)((char*)d_ws + WS_KC);
    __bf16*   Vt = (__bf16*)((char*)d_ws + WS_VT);
    unsigned* mb = (unsigned*)((char*)d_ws + WS_MB);

    mask_kernel<<<(SEQ + 255) / 256, 256, 0, stream>>>(rm, mb);
    convert_kernel<<<(HEADS * NKEYS * DHEAD) / 256, 256, 0, stream>>>(k, v, rk, rv, Kc, Vt);

    dim3 grid(SEQ / 128, HEADS);
    attn_kernel<<<grid, 256, SMEM_BYTES, stream>>>(q, Kc, Vt, mb, (float*)d_out);
}
